// local_conv_1219770712072
// MI455X (gfx1250) — matrix operand order repaired and checked by reading, not compiled, not run on hardware
//
#include <hip/hip_runtime.h>

// ---------------------------------------------------------------------------
// CDNA5 (gfx1250) bf16 WMMA implementation of the attention+inception block.
// 128x64 output tile per 128-thread block, 64-deep K per stage,
// 16 x v_wmma_f32_16x16x32_bf16 per stage (2 A-frags x 4 B-frags x 2 K),
// double-buffered LDS with GLOBAL_LOAD_ASYNC_TO_LDS_B128 prefetch of the
// A (weight) tile overlapping the WMMA phase; one barrier per K-stage.
// ---------------------------------------------------------------------------

typedef __attribute__((ext_vector_type(16))) __bf16 v16bf;
typedef __attribute__((ext_vector_type(8)))  __bf16 v8bf;
typedef __attribute__((ext_vector_type(8)))  float  v8f;

#define STR  72   // LDS pitch in bf16 elems (144B): 16B-aligned, conflict-free
#define PADP 64   // padded P stride for bf16 activation tensors (49 -> 64)

// Async global -> LDS copy, 16B per lane, tracked by ASYNCcnt.
__device__ __forceinline__ void async_b128(unsigned lds_off, const void* gaddr) {
  asm volatile("global_load_async_to_lds_b128 %0, %1, off"
               :: "v"(lds_off), "v"((unsigned long long)(uintptr_t)gaddr)
               : "memory");
}
__device__ __forceinline__ void wait_async0() {
  asm volatile("s_wait_asynccnt 0" ::: "memory");
}

// A-fragment per ISA 16-bit 16x32 layout.
__device__ __forceinline__ v16bf frag_a(const __bf16* lds, int rowBase, int s, int lane) {
  int row = rowBase + (lane & 15);
  int kb  = s * 32 + ((lane >> 4) << 3);
  const __bf16* p = lds + row * STR + kb;
  v8bf lo = *(const v8bf*)(p);        // K = kb..kb+7
  v8bf hi = *(const v8bf*)(p + 16);   // K = kb+16..kb+23
  v16bf r;
#pragma unroll
  for (int i = 0; i < 8; ++i) { r[i] = lo[i]; r[8 + i] = hi[i]; }
  return r;
}

// B-fragment; B stored transposed in LDS as [p][k].
__device__ __forceinline__ v16bf frag_b(const __bf16* lds, int j, int s, int lane) {
  int col = j * 16 + (lane & 15);
  int kb  = s * 32 + ((lane >> 4) << 3);
  const __bf16* p = lds + col * STR + kb;
  v8bf lo = *(const v8bf*)(p);
  v8bf hi = *(const v8bf*)(p + 16);
  v16bf r;
#pragma unroll
  for (int i = 0; i < 8; ++i) { r[i] = lo[i]; r[8 + i] = hi[i]; }
  return r;
}

#define WMMA_BF16(A_, B_, C_) \
  __builtin_amdgcn_wmma_f32_16x16x32_bf16(false, A_, false, B_, (short)0, C_, false, false)

// Batched GEMM (+conv shift for KDIM=3/5).
//   Out[n][m][p] = sum_kk A[n*aBatch + m*aSM + kk*aSK] * Xshift[n][kk][p]
//   kk -> (tap = kk>>cinShift, c = kk&mask); X is [C][PADP] per n, zero-padded.
template <int KDIM, bool AVEC>
__global__ __launch_bounds__(128) void gemm_bf16_kernel(
    const __bf16* __restrict__ A, long long aBatch, long long aSM, long long aSK,
    const __bf16* __restrict__ X, long long xBatch, int cinShift,
    int K, int Mtot,
    float* __restrict__ outF, long long outFB,
    __bf16* __restrict__ outB, long long outBB,
    int relu,
    const float* __restrict__ fuseG, const float* __restrict__ fuseI, long long fuseB)
{
  __shared__ __align__(16) __bf16 Alds[2][128 * STR];
  __shared__ __align__(16) __bf16 Blds[2][64 * STR];

  const int n    = blockIdx.x;
  const int m0   = blockIdx.y * 128;
  const int t    = threadIdx.x;
  const int lane = t & 31;
  const int wave = t >> 5;
  const __bf16* An = A + (long long)n * aBatch;
  const __bf16* Xn = X + (long long)n * xBatch;
  const int pad     = KDIM / 2;
  const int cinMask = (1 << cinShift) - 1;

  v8f acc[8];
#pragma unroll
  for (int i = 0; i < 8; ++i) acc[i] = (v8f){};

  auto fill_tile = [&](int k0, int buf) {
    __bf16* Ab = Alds[buf];
    __bf16* Bb = Blds[buf];
    // ---- A tile: 128 rows x 64 K (1024 16B chunks / 128 threads) ----
#pragma unroll
    for (int i = 0; i < 8; ++i) {
      int q   = i * 128 + t;
      int row = q >> 3;
      int ko  = (q & 7) * 8;
      if (AVEC) {
        async_b128((unsigned)(uintptr_t)(Ab + row * STR + ko),
                   An + (long long)(m0 + row) * aSM + (k0 + ko));
      } else {
        const __bf16* src = An + (long long)(m0 + row) * aSM + (long long)(k0 + ko) * aSK;
#pragma unroll
        for (int j2 = 0; j2 < 8; ++j2)
          Ab[row * STR + ko + j2] = src[(long long)j2 * aSK];
      }
    }
    // ---- B tile: 64 K-rows x 64 P cols, stored transposed [p][k] ----
    if (KDIM == 1) {
#pragma unroll
      for (int i = 0; i < 4; ++i) {
        int q    = i * 128 + t;
        int krow = q >> 3;
        int po   = (q & 7) * 8;
        v8bf v = *(const v8bf*)(Xn + (long long)(k0 + krow) * PADP + po);
#pragma unroll
        for (int j2 = 0; j2 < 8; ++j2)
          Bb[(po + j2) * STR + krow] = v[j2];
      }
    } else {
      // each thread owns a fixed k; all k-dependent math hoisted out of p loop
      int k  = t & 63;
      int pb = t >> 6;                // 0 or 1
      int kk = k0 + k;
      int c   = kk & cinMask;
      int tap = kk >> cinShift;
      int kh = tap / KDIM, kw = tap - kh * KDIM;
      int doff = (kh - pad) * 7 + (kw - pad);
      const __bf16* xc = Xn + (long long)c * PADP;
#pragma unroll 4
      for (int i = 0; i < 32; ++i) {
        int p = i * 2 + pb;
        int h = p / 7, w = p - h * 7;
        int hh = h + kh - pad, ww = w + kw - pad;
        __bf16 v = (__bf16)0.0f;
        if (hh >= 0 && hh < 7 && ww >= 0 && ww < 7) v = xc[p + doff];
        Bb[p * STR + k] = v;
      }
    }
  };

  const int nIter = K >> 6;
  fill_tile(0, 0);
  wait_async0();
  __syncthreads();

  for (int it = 0; it < nIter; ++it) {
    int buf = it & 1;
    if (it + 1 < nIter) fill_tile((it + 1) << 6, buf ^ 1);  // overlaps WMMA below
    const __bf16* Ab = Alds[buf];
    const __bf16* Bb = Blds[buf];
#pragma unroll
    for (int s = 0; s < 2; ++s) {
      v16bf a0 = frag_a(Ab, wave * 32,      s, lane);
      v16bf a1 = frag_a(Ab, wave * 32 + 16, s, lane);
#pragma unroll
      for (int j = 0; j < 4; ++j) {
        v16bf b = frag_b(Bb, j, s, lane);
        acc[j]     = WMMA_BF16(a0, b, acc[j]);
        acc[4 + j] = WMMA_BF16(a1, b, acc[4 + j]);
      }
    }
    wait_async0();      // next tile's async copies done before anyone reads them
    __syncthreads();
  }

  // ---- epilogue (f32 C/D layout: VGPR r -> row, col = lane&15) ----
  const int colL = lane & 15;
#pragma unroll
  for (int rg = 0; rg < 2; ++rg) {
    int rbase = m0 + wave * 32 + rg * 16 + ((lane >> 4) << 3);
#pragma unroll
    for (int j = 0; j < 4; ++j) {
      int p = j * 16 + colL;
      v8f av = acc[rg * 4 + j];
#pragma unroll
      for (int r = 0; r < 8; ++r) {
        int m = rbase + r;
        if (m >= Mtot) continue;
        float v = av[r];
        if (relu) v = v > 0.0f ? v : 0.0f;
        if (outB) outB[(long long)n * outBB + (long long)m * PADP + p] = (__bf16)v;
        if (outF && p < 49) {
          long long off = (long long)m * 49 + p;
          if (fuseG) {
            long long fo = (long long)n * fuseB + off;
            v = v + fuseG[fo] * fuseI[fo];
          }
          outF[(long long)n * outFB + off] = v;
        }
      }
    }
  }
}

// softmax over l (64) of dot/T per (n,p); zero-fills bf16 padding columns.
__global__ void softmax_attn_kernel(const float* __restrict__ dot,
                                    float* __restrict__ attF,
                                    __bf16* __restrict__ attB)
{
  int i = blockIdx.x * blockDim.x + threadIdx.x;   // over N * PADP
  if (i >= 256 * PADP) return;
  int n = i >> 6, p = i & 63;
  __bf16* ab = attB + (long long)n * 64 * PADP + p;
  if (p >= 49) {
    for (int l = 0; l < 64; ++l) ab[(long long)l * PADP] = (__bf16)0.0f;
    return;
  }
  const float* d = dot + (long long)n * 64 * 49 + p;
  float* af = attF + (long long)n * 64 * 49 + p;
  const float invT = 1.0f / 7.0f;
  float mx = -3.4e38f;
  for (int l = 0; l < 64; ++l) mx = fmaxf(mx, d[l * 49] * invT);
  float s = 0.0f;
  for (int l = 0; l < 64; ++l) s += __expf(d[l * 49] * invT - mx);
  float inv = 1.0f / s;
  for (int l = 0; l < 64; ++l) {
    float a = __expf(d[l * 49] * invT - mx) * inv;
    af[(long long)l * 49] = a;
    ab[(long long)l * PADP] = (__bf16)a;
  }
}

__global__ void cvt_kernel(const float* __restrict__ in, __bf16* __restrict__ out,
                           long long nElem) {
  long long i = (long long)blockIdx.x * blockDim.x + threadIdx.x;
  if (i < nElem) out[i] = (__bf16)in[i];
}

// compact [rows,49] f32 -> padded [rows,64] bf16 (padding = 0)
__global__ void cvt_pad_kernel(const float* __restrict__ in, __bf16* __restrict__ out,
                               long long nRows) {
  long long i = (long long)blockIdx.x * blockDim.x + threadIdx.x;
  if (i >= nRows * PADP) return;
  long long row = i >> 6;
  int p = (int)(i & 63);
  out[i] = (p < 49) ? (__bf16)in[row * 49 + p] : (__bf16)0.0f;
}

// [O,C,kh,kw] f32 -> [O, tap, C] bf16 so conv GEMM K = tap*C + c is contiguous.
__global__ void cvt_perm_kernel(const float* __restrict__ in, __bf16* __restrict__ out,
                                int O, int C, int T) {
  long long i = (long long)blockIdx.x * blockDim.x + threadIdx.x;
  long long total = (long long)O * C * T;
  if (i >= total) return;
  int o   = (int)(i / ((long long)C * T));
  int rem = (int)(i - (long long)o * C * T);
  int c  = rem / T;
  int tp = rem - c * T;
  out[((long long)o * T + tp) * C + c] = (__bf16)in[i];
}

extern "C" void kernel_launch(void* const* d_in, const int* in_sizes, int n_in,
                              void* d_out, int out_size, void* d_ws, size_t ws_size,
                              hipStream_t stream)
{
  (void)in_sizes; (void)n_in; (void)out_size; (void)ws_size;
  const int N = 256;
  const long long NP2048  = (long long)N * 2048 * 49;   // compact f32 plane
  const long long NP64    = (long long)N * 64 * 49;
  const long long XPB     = 2048LL * PADP;              // padded batch stride

  const float* img  = (const float*)d_in[0];
  const float* text = (const float*)d_in[1];
  const float* Wimg = (const float*)d_in[2];
  const float* wF[10];
  for (int i = 0; i < 10; ++i) wF[i] = (const float*)d_in[3 + i];
  // wF: g_w1 g_w2a g_w2b g_w3a g_w3b b_w1 b_w2a b_w2b b_w3a b_w3b

  float* out  = (float*)d_out;
  float* attF = out + NP2048;          // second tuple output: atten [256,64,49]

  // ---- workspace bump allocator ----
  char* cur = (char*)d_ws;
  auto alloc = [&](long long elems, int esz) -> void* {
    void* p = (void*)cur;
    long long bytes = elems * (long long)esz;
    cur += (bytes + 255) & ~(long long)255;
    return p;
  };
  __bf16* imgbf  = (__bf16*)alloc((long long)N * 2048 * PADP, 2);
  __bf16* textbf = (__bf16*)alloc((long long)N * 64 * 1024, 2);
  __bf16* wimgbf = (__bf16*)alloc(1024LL * 2048, 2);
  __bf16* xbf    = (__bf16*)alloc((long long)N * 2048 * PADP, 2);
  float*  dotF   = (float*)alloc(NP64, 4);
  __bf16* attB   = (__bf16*)alloc((long long)N * 64 * PADP, 2);
  __bf16* y2     = (__bf16*)alloc((long long)N * 512 * PADP, 2);
  __bf16* y3     = (__bf16*)alloc((long long)N * 256 * PADP, 2);
  float*  gbuf   = (float*)alloc(NP2048, 4);
  const long long wElems[10] = {512LL*2048, 512LL*2048, 1024LL*512*9, 256LL*2048, 512LL*256*25,
                                512LL*2048, 512LL*2048, 1024LL*512*9, 256LL*2048, 512LL*256*25};
  __bf16* wB[10];
  for (int i = 0; i < 10; ++i) wB[i] = (__bf16*)alloc(wElems[i], 2);

  auto cvt = [&](const float* a, __bf16* b, long long nE) {
    cvt_kernel<<<dim3((unsigned)((nE + 255) / 256)), dim3(256), 0, stream>>>(a, b, nE);
  };
  auto cvtp = [&](const float* a, __bf16* b, int O, int C, int T) {
    long long nE = (long long)O * C * T;
    cvt_perm_kernel<<<dim3((unsigned)((nE + 255) / 256)), dim3(256), 0, stream>>>(a, b, O, C, T);
  };

  {
    long long nE = (long long)N * 2048 * PADP;
    cvt_pad_kernel<<<dim3((unsigned)((nE + 255) / 256)), dim3(256), 0, stream>>>(
        img, imgbf, (long long)N * 2048);
  }
  cvt(text, textbf, (long long)N * 64 * 1024);
  cvt(Wimg, wimgbf, 1024LL * 2048);
  cvt(wF[0], wB[0], wElems[0]);
  cvt(wF[1], wB[1], wElems[1]);
  cvtp(wF[2], wB[2], 1024, 512, 9);
  cvt(wF[3], wB[3], wElems[3]);
  cvtp(wF[4], wB[4], 512, 256, 25);
  cvt(wF[5], wB[5], wElems[5]);
  cvt(wF[6], wB[6], wElems[6]);
  cvtp(wF[7], wB[7], 1024, 512, 9);
  cvt(wF[8], wB[8], wElems[8]);
  cvtp(wF[9], wB[9], 512, 256, 25);

  const dim3 blk(128);
  // 1) img_embed = Wimg x img -> xbf channels [0,1024)
  gemm_bf16_kernel<1, true><<<dim3(N, 8), blk, 0, stream>>>(
      wimgbf, 0, 2048, 1, imgbf, XPB, 0, 2048, 1024,
      nullptr, 0, xbf, XPB, 0, nullptr, nullptr, 0);
  // 2) dot[n,l,p] = text[n] x img_embed[n]
  gemm_bf16_kernel<1, true><<<dim3(N, 1), blk, 0, stream>>>(
      textbf, 64LL * 1024, 1024, 1, xbf, XPB, 0, 1024, 64,
      dotF, 64LL * 49, nullptr, 0, 0, nullptr, nullptr, 0);
  // 3) softmax over l
  softmax_attn_kernel<<<dim3((N * PADP + 255) / 256), dim3(256), 0, stream>>>(dotF, attF, attB);
  // 4) sentence_cat = text^T x atten -> xbf channels [1024,2048)
  gemm_bf16_kernel<1, false><<<dim3(N, 8), blk, 0, stream>>>(
      textbf, 64LL * 1024, 1, 1024, attB, 64LL * PADP, 0, 64, 1024,
      nullptr, 0, xbf + 1024LL * PADP, XPB, 0, nullptr, nullptr, 0);

  // ---- gamma inception -> gbuf (compact f32) ----
  gemm_bf16_kernel<1, true><<<dim3(N, 4), blk, 0, stream>>>(
      wB[0], 0, 2048, 1, xbf, XPB, 0, 2048, 512,
      gbuf, 2048LL * 49, nullptr, 0, 0, nullptr, nullptr, 0);
  gemm_bf16_kernel<1, true><<<dim3(N, 4), blk, 0, stream>>>(
      wB[1], 0, 2048, 1, xbf, XPB, 0, 2048, 512,
      nullptr, 0, y2, 512LL * PADP, 1, nullptr, nullptr, 0);
  gemm_bf16_kernel<3, true><<<dim3(N, 8), blk, 0, stream>>>(
      wB[2], 0, 4608, 1, y2, 512LL * PADP, 9, 4608, 1024,
      gbuf + 512LL * 49, 2048LL * 49, nullptr, 0, 0, nullptr, nullptr, 0);
  gemm_bf16_kernel<1, true><<<dim3(N, 2), blk, 0, stream>>>(
      wB[3], 0, 2048, 1, xbf, XPB, 0, 2048, 256,
      nullptr, 0, y3, 256LL * PADP, 1, nullptr, nullptr, 0);
  gemm_bf16_kernel<5, true><<<dim3(N, 4), blk, 0, stream>>>(
      wB[4], 0, 6400, 1, y3, 256LL * PADP, 8, 6400, 512,
      gbuf + 1536LL * 49, 2048LL * 49, nullptr, 0, 0, nullptr, nullptr, 0);

  // ---- beta inception, fused out = gamma*img + beta ----
  gemm_bf16_kernel<1, true><<<dim3(N, 4), blk, 0, stream>>>(
      wB[5], 0, 2048, 1, xbf, XPB, 0, 2048, 512,
      out, 2048LL * 49, nullptr, 0, 0, gbuf, img, 2048LL * 49);
  gemm_bf16_kernel<1, true><<<dim3(N, 4), blk, 0, stream>>>(
      wB[6], 0, 2048, 1, xbf, XPB, 0, 2048, 512,
      nullptr, 0, y2, 512LL * PADP, 1, nullptr, nullptr, 0);
  gemm_bf16_kernel<3, true><<<dim3(N, 8), blk, 0, stream>>>(
      wB[7], 0, 4608, 1, y2, 512LL * PADP, 9, 4608, 1024,
      out + 512LL * 49, 2048LL * 49, nullptr, 0, 0,
      gbuf + 512LL * 49, img + 512LL * 49, 2048LL * 49);
  gemm_bf16_kernel<1, true><<<dim3(N, 2), blk, 0, stream>>>(
      wB[8], 0, 2048, 1, xbf, XPB, 0, 2048, 256,
      nullptr, 0, y3, 256LL * PADP, 1, nullptr, nullptr, 0);
  gemm_bf16_kernel<5, true><<<dim3(N, 4), blk, 0, stream>>>(
      wB[9], 0, 6400, 1, y3, 256LL * PADP, 8, 6400, 512,
      out + 1536LL * 49, 2048LL * 49, nullptr, 0, 0,
      gbuf + 1536LL * 49, img + 1536LL * 49, 2048LL * 49);
}
